// MemorySelfAttention_30554397344331
// MI455X (gfx1250) — matrix operand order repaired and checked by reading, not compiled, not run on hardware
//
#include <hip/hip_runtime.h>

// ---------------------------------------------------------------------------
// MI455X / gfx1250 (CDNA5, wave32) implementation of MemorySelfAttention.
// bf16 WMMA (v_wmma_f32_16x16x32_bf16) for all GEMMs + flash attention.
// GEMM tiles are filled with CDNA5 async global->LDS DMA (ASYNCcnt-tracked),
// double-buffered in LDS, last iteration peeled so the hot loop is branchless.
// ---------------------------------------------------------------------------

typedef __attribute__((ext_vector_type(16))) __bf16 v16bf;
typedef __attribute__((ext_vector_type(8)))  __bf16 v8bf;
typedef __attribute__((ext_vector_type(8)))  float  v8f;

union Frag16 { v16bf v; v8bf h[2]; unsigned short s[16]; };

#if __has_builtin(__builtin_amdgcn_exp2f)
#define EXP2F(x) __builtin_amdgcn_exp2f(x)
#else
#define EXP2F(x) exp2f(x)
#endif

#define L2E 1.44269504088896340736f
#define LOG2_THETA 13.287712379549449f   // log2(10000)
#define NEG_BIG (-3.0e38f)

// Problem dims
#define TN   2048
#define SN   512
#define LN   1024
#define CN   2048
#define HN   16
#define HKN  4
#define DN   128
#define SEQN 2560          // S + T
#define TOTN 3584          // L + S + T
#define KVN  512           // Hk * D

__device__ __forceinline__ unsigned short f2bf(float f) {
    unsigned int u = __float_as_uint(f);
    u += 0x7FFFu + ((u >> 16) & 1u);        // round-to-nearest-even
    return (unsigned short)(u >> 16);
}

// Issue one per-lane 16B async global->LDS copy (CDNA5 async DMA path).
__device__ __forceinline__ void async_b128(unsigned lds_addr, const void* gaddr) {
    asm volatile("global_load_async_to_lds_b128 %0, %1, off"
                 :: "v"(lds_addr), "v"((unsigned long long)(uintptr_t)gaddr)
                 : "memory");
}

// ---------------------------------------------------------------------------
// fp32 -> bf16 convert (vectorized x4)
// ---------------------------------------------------------------------------
__global__ __launch_bounds__(256) void k_cvt_bf16(const float* __restrict__ src,
                                                  unsigned short* __restrict__ dst,
                                                  int n4) {
    int t = blockIdx.x * 256 + threadIdx.x;
    if (t < n4) {
        float4 f = ((const float4*)src)[t];
        ushort4 o;
        o.x = f2bf(f.x); o.y = f2bf(f.y); o.z = f2bf(f.z); o.w = f2bf(f.w);
        ((ushort4*)dst)[t] = o;
    }
}

// ---------------------------------------------------------------------------
// fp32 W[K][N] -> bf16 W^T[N][K], 64x64 LDS tile transpose (coalesced both ways)
// grid: (Ndim/64, Kdim/64)
// ---------------------------------------------------------------------------
__global__ __launch_bounds__(256) void k_cvt_tr(const float* __restrict__ S,
                                                unsigned short* __restrict__ D,
                                                int Kdim, int Ndim) {
    __shared__ __align__(16) float tile[64][65];
    const int nb = blockIdx.x * 64;
    const int kb = blockIdx.y * 64;
    const int tid = threadIdx.x;
    #pragma unroll
    for (int i = 0; i < 4; ++i) {
        int r = (tid >> 4) + i * 16;          // 0..63 (k)
        int c = (tid & 15) * 4;               // 0..60 (n)
        float4 f = *(const float4*)&S[(size_t)(kb + r) * Ndim + nb + c];
        tile[r][c] = f.x; tile[r][c + 1] = f.y; tile[r][c + 2] = f.z; tile[r][c + 3] = f.w;
    }
    __syncthreads();
    const int n = tid >> 2;                   // 0..63
    const int ks = (tid & 3) * 16;            // 0,16,32,48
    #pragma unroll
    for (int i = 0; i < 16; i += 4) {
        ushort4 o;
        o.x = f2bf(tile[ks + i][n]);
        o.y = f2bf(tile[ks + i + 1][n]);
        o.z = f2bf(tile[ks + i + 2][n]);
        o.w = f2bf(tile[ks + i + 3][n]);
        *(ushort4*)&D[(size_t)(nb + n) * Kdim + kb + ks + i] = o;
    }
}

// ---------------------------------------------------------------------------
// bf16 GEMM: C(f32)[M][N] = A[M][K] * Bt[N][K]^T   (Bt is pre-transposed)
// Block: 256 thr (8 waves), tile 128x128x32, wave -> 64x32 (4x2 WMMA tiles).
// Double-buffered LDS filled by async global->LDS b128 DMA.
// ---------------------------------------------------------------------------
#define BM 128
#define BN 128
#define BK 32

__global__ __launch_bounds__(256) void k_gemm_bf16(const unsigned short* __restrict__ A,
                                                   const unsigned short* __restrict__ Bt,
                                                   float* __restrict__ C,
                                                   int M, int N, int K) {
    __shared__ __align__(16) unsigned short As[2][BM][BK];   // 2 x 8KB
    __shared__ __align__(16) unsigned short Bs[2][BN][BK];   // 2 x 8KB

    const int bn   = blockIdx.x * BN;
    const int bm   = blockIdx.y * BM;
    const int tid  = threadIdx.x;
    const int lane = tid & 31;
    const int w    = tid >> 5;
    const int wm   = (w >> 2) * 64;       // 0 / 64
    const int wn   = (w & 3) * 32;        // 0 / 32 / 64 / 96
    const int row16 = lane & 15;
    const int aoff  = (lane & 16) ? 16 : 0;   // A-layout K offset
    const int boff  = (lane & 16) ? 16 : 0;   // B-layout K offset
    const int mhalf = (lane >> 4) * 8;        // C-layout row half

    // Per-thread fill slots: rows fr and fr+64, 8-element column chunk fc.
    const int fr = tid >> 2;              // 0..63
    const int fc = (tid & 3) * 8;         // 0,8,16,24
    const unsigned short* gA0 = A  + (size_t)(bm + fr) * K + fc;
    const unsigned short* gA1 = A  + (size_t)(bm + fr + 64) * K + fc;
    const unsigned short* gB0 = Bt + (size_t)(bn + fr) * K + fc;
    const unsigned short* gB1 = Bt + (size_t)(bn + fr + 64) * K + fc;
    const unsigned ldsA = (unsigned)(uintptr_t)&As[0][fr][fc];
    const unsigned ldsB = (unsigned)(uintptr_t)&Bs[0][fr][fc];
    const unsigned rowB = 64 * BK * 2;    // 64 rows * 64B
    const unsigned bufB = BM * BK * 2;    // 8KB buffer stride

    v8f acc[4][2] = {};

    auto compute = [&](int buf) {
        Frag16 af[4], bf_[2];
        #pragma unroll
        for (int i = 0; i < 4; ++i) {
            int ar = wm + 16 * i + row16;
            af[i].h[0] = *(const v8bf*)&As[buf][ar][aoff];
            af[i].h[1] = *(const v8bf*)&As[buf][ar][aoff + 8];
        }
        #pragma unroll
        for (int jt = 0; jt < 2; ++jt) {
            int br = wn + 16 * jt + row16;
            bf_[jt].h[0] = *(const v8bf*)&Bs[buf][br][boff];
            bf_[jt].h[1] = *(const v8bf*)&Bs[buf][br][boff + 8];
        }
        #pragma unroll
        for (int i = 0; i < 4; ++i)
            #pragma unroll
            for (int jt = 0; jt < 2; ++jt)
                acc[i][jt] = __builtin_amdgcn_wmma_f32_16x16x32_bf16(
                    false, af[i].v, false, bf_[jt].v, (short)0, acc[i][jt], false, false);
    };

    // prologue: tile 0 -> buffer 0
    async_b128(ldsA,        gA0);
    async_b128(ldsA + rowB, gA1);
    async_b128(ldsB,        gB0);
    async_b128(ldsB + rowB, gB1);

    const int nk = K / BK;
    for (int it = 0; it < nk - 1; ++it) {
        const int buf = it & 1;
        const int k0  = (it + 1) * BK;
        const unsigned bo = (buf ^ 1) * bufB;
        async_b128(ldsA + bo,        gA0 + k0);
        async_b128(ldsA + bo + rowB, gA1 + k0);
        async_b128(ldsB + bo,        gB0 + k0);
        async_b128(ldsB + bo + rowB, gB1 + k0);
        asm volatile("s_wait_asynccnt 0x4" ::: "memory");   // tile `it` landed
        __syncthreads();
        compute(buf);
        __syncthreads();    // protect LDS WAR for next async fill
    }
    asm volatile("s_wait_asynccnt 0x0" ::: "memory");       // last tile landed
    __syncthreads();
    compute((nk - 1) & 1);

    #pragma unroll
    for (int i = 0; i < 4; ++i)
        #pragma unroll
        for (int jt = 0; jt < 2; ++jt)
            #pragma unroll
            for (int r = 0; r < 8; ++r) {
                int m = bm + wm + 16 * i + mhalf + r;
                int n = bn + wn + 16 * jt + row16;
                C[(size_t)m * N + n] = acc[i][jt][r];
            }
}

// ---------------------------------------------------------------------------
// RoPE on Q projection; fold 1/sqrt(D); write head-major bf16 Qb[h][row][d]
// rope position = 512 + row (query global pos 1536+row, minus L=1024)
// ---------------------------------------------------------------------------
__global__ __launch_bounds__(256) void k_rope_q(const float* __restrict__ Qf,
                                                unsigned short* __restrict__ Qb) {
    int t = blockIdx.x * 256 + threadIdx.x;       // TN*HN*64 threads
    int i   = t & 63;
    int h   = (t >> 6) & 15;
    int row = t >> 10;
    float freq = EXP2F(-(float)i * (LOG2_THETA / 64.0f));
    float ang  = (float)(512 + row) * freq;
    float c = cosf(ang), s = sinf(ang);
    size_t q = (size_t)row * CN + h * DN + 2 * i;
    float x1 = Qf[q], x2 = Qf[q + 1];
    const float sc = 0.08838834764831845f;        // 1/sqrt(128)
    size_t o = ((size_t)h * TN + row) * DN + 2 * i;
    Qb[o]     = f2bf((x1 * c - x2 * s) * sc);
    Qb[o + 1] = f2bf((x1 * s + x2 * c) * sc);
}

// ---------------------------------------------------------------------------
// Build Kb[kh][j][d] bf16: long_k rows (no rope) + projected rows (rope)
// ---------------------------------------------------------------------------
__global__ __launch_bounds__(256) void k_build_k(const float* __restrict__ long_k,
                                                 const float* __restrict__ Kf,
                                                 unsigned short* __restrict__ Kb) {
    int t = blockIdx.x * 256 + threadIdx.x;       // HKN*TOTN*64 threads
    int i = t & 63;
    int j = (t >> 6) % TOTN;
    int h = t / (TOTN * 64);
    float x1, x2;
    if (j < LN) {
        size_t s = ((size_t)j * HKN + h) * DN + 2 * i;
        x1 = long_k[s]; x2 = long_k[s + 1];
    } else {
        int pos = j - LN;
        size_t s = (size_t)pos * KVN + h * DN + 2 * i;
        float a = Kf[s], b = Kf[s + 1];
        float freq = EXP2F(-(float)i * (LOG2_THETA / 64.0f));
        float ang  = (float)pos * freq;
        float c = cosf(ang), sn = sinf(ang);
        x1 = a * c - b * sn;
        x2 = a * sn + b * c;
    }
    size_t o = ((size_t)h * TOTN + j) * DN + 2 * i;
    Kb[o] = f2bf(x1); Kb[o + 1] = f2bf(x2);
}

// ---------------------------------------------------------------------------
// Build transposed Vt[kh][d][j] bf16
// ---------------------------------------------------------------------------
__global__ __launch_bounds__(256) void k_build_v(const float* __restrict__ long_v,
                                                 const float* __restrict__ Vf,
                                                 unsigned short* __restrict__ Vt) {
    int t = blockIdx.x * 256 + threadIdx.x;       // HKN*DN*TOTN threads
    int j = t % TOTN;
    int d = (t / TOTN) & (DN - 1);
    int h = t / (TOTN * DN);
    float v = (j < LN) ? long_v[((size_t)j * HKN + h) * DN + d]
                       : Vf[(size_t)(j - LN) * KVN + h * DN + d];
    Vt[((size_t)h * DN + d) * TOTN + j] = f2bf(v);
}

// ---------------------------------------------------------------------------
// Flash attention. Block = (head, 128-query tile), 8 waves, wave = 16 rows.
// Streams 32-key tiles: S = Q K^T (8 wmma), online softmax, P via LDS
// (C-layout -> A-layout), O += P V (8 wmma). Writes Yb[row][h*128+d] bf16.
// ---------------------------------------------------------------------------
__global__ __launch_bounds__(256) void k_attn(const unsigned short* __restrict__ Qb,
                                              const unsigned short* __restrict__ Kb,
                                              const unsigned short* __restrict__ Vt,
                                              unsigned short* __restrict__ Yb) {
    __shared__ __align__(16) unsigned short Pl[8][16][32];

    const int h    = blockIdx.x;          // 0..15
    const int qt   = blockIdx.y;          // 0..15
    const int tid  = threadIdx.x;
    const int lane = tid & 31;
    const int w    = tid >> 5;
    const int kh   = h >> 2;              // GQA: 4 q-heads per kv-head
    const int row16 = lane & 15;
    const int aoff  = (lane & 16) ? 16 : 0;
    const int boff  = (lane & 16) ? 16 : 0;
    const int mhalf = (lane >> 4) * 8;

    const int qr0 = qt * 128 + w * 16;    // this wave's query rows (local)
    const unsigned short* Qh = Qb + ((size_t)h * TN + qr0) * DN;
    const unsigned short* Kh = Kb + (size_t)kh * TOTN * DN;
    const unsigned short* Vh = Vt + (size_t)kh * DN * TOTN;

    // Q fragments (A-layout), 4 K-chunks of 32 along d
    Frag16 qf[4];
    #pragma unroll
    for (int dc = 0; dc < 4; ++dc) {
        const unsigned short* p = Qh + (size_t)row16 * DN + dc * 32 + aoff;
        qf[dc].h[0] = *(const v8bf*)&p[0];
        qf[dc].h[1] = *(const v8bf*)&p[8];
    }

    v8f o[8] = {};
    float mx[8], l[8];
    #pragma unroll
    for (int r = 0; r < 8; ++r) { mx[r] = NEG_BIG; l[r] = 0.0f; }

    const int gi_base = LN + SN + qr0;    // global pos of row m=0
    const int gimax   = gi_base + 15;

    for (int j0 = 0; j0 <= gimax; j0 += 32) {
        // prefetch next key/value tiles (gfx1250 global_prefetch_b8)
        __builtin_prefetch(Kh + (size_t)(j0 + 32 + row16) * DN, 0, 0);
        __builtin_prefetch(Vh + (size_t)(lane) * TOTN + j0 + 32, 0, 0);
        // ---- scores: S(16x32) as two 16x16 fragments
        v8f s0 = {}, s1 = {};
        #pragma unroll
        for (int dc = 0; dc < 4; ++dc) {
            Frag16 kf0, kf1;
            const unsigned short* p0 = Kh + (size_t)(j0 + row16) * DN + dc * 32 + boff;
            const unsigned short* p1 = p0 + 16 * DN;
            kf0.h[0] = *(const v8bf*)&p0[0];  kf0.h[1] = *(const v8bf*)&p0[8];
            kf1.h[0] = *(const v8bf*)&p1[0];  kf1.h[1] = *(const v8bf*)&p1[8];
            s0 = __builtin_amdgcn_wmma_f32_16x16x32_bf16(false, qf[dc].v, false, kf0.v,
                                                         (short)0, s0, false, false);
            s1 = __builtin_amdgcn_wmma_f32_16x16x32_bf16(false, qf[dc].v, false, kf1.v,
                                                         (short)0, s1, false, false);
        }
        // ---- causal mask
        #pragma unroll
        for (int r = 0; r < 8; ++r) {
            int gi = gi_base + mhalf + r;
            s0[r] = ((j0 + row16)      > gi) ? NEG_BIG : s0[r];
            s1[r] = ((j0 + 16 + row16) > gi) ? NEG_BIG : s1[r];
        }
        // ---- online softmax (rows live in 16-lane halves of C-layout)
        #pragma unroll
        for (int r = 0; r < 8; ++r) {
            float t = fmaxf(s0[r], s1[r]);
            t = fmaxf(t, __shfl_xor(t, 1, 32));
            t = fmaxf(t, __shfl_xor(t, 2, 32));
            t = fmaxf(t, __shfl_xor(t, 4, 32));
            t = fmaxf(t, __shfl_xor(t, 8, 32));
            float mnew  = fmaxf(mx[r], t);
            float alpha = EXP2F((mx[r] - mnew) * L2E);
            float p0 = EXP2F((s0[r] - mnew) * L2E);
            float p1 = EXP2F((s1[r] - mnew) * L2E);
            s0[r] = p0; s1[r] = p1;
            float rs = p0 + p1;
            rs += __shfl_xor(rs, 1, 32);
            rs += __shfl_xor(rs, 2, 32);
            rs += __shfl_xor(rs, 4, 32);
            rs += __shfl_xor(rs, 8, 32);
            l[r]  = l[r] * alpha + rs;
            mx[r] = mnew;
            #pragma unroll
            for (int dt = 0; dt < 8; ++dt) o[dt][r] *= alpha;
        }
        // ---- P: C-layout -> A-layout through per-wave LDS
        #pragma unroll
        for (int r = 0; r < 8; ++r) {
            Pl[w][mhalf + r][row16]      = f2bf(s0[r]);
            Pl[w][mhalf + r][row16 + 16] = f2bf(s1[r]);
        }
        asm volatile("s_wait_dscnt 0x0" ::: "memory");   // wave-local LDS RAW
        Frag16 pf;
        pf.h[0] = *(const v8bf*)&Pl[w][row16][aoff];
        pf.h[1] = *(const v8bf*)&Pl[w][row16][aoff + 8];
        // ---- O += P * V  (V^T rows are contiguous in j)
        #pragma unroll
        for (int dt = 0; dt < 8; ++dt) {
            Frag16 vf;
            const unsigned short* p = Vh + (size_t)(dt * 16 + row16) * TOTN + j0 + boff;
            vf.h[0] = *(const v8bf*)&p[0];
            vf.h[1] = *(const v8bf*)&p[8];
            o[dt] = __builtin_amdgcn_wmma_f32_16x16x32_bf16(false, pf.v, false, vf.v,
                                                            (short)0, o[dt], false, false);
        }
    }

    // ---- normalize and store bf16 Y[row][h*128 + d]
    #pragma unroll
    for (int r = 0; r < 8; ++r) l[r] = 1.0f / l[r];
    #pragma unroll
    for (int dt = 0; dt < 8; ++dt)
        #pragma unroll
        for (int r = 0; r < 8; ++r) {
            int m = qr0 + mhalf + r;
            int n = h * DN + dt * 16 + row16;
            Yb[(size_t)m * CN + n] = f2bf(o[dt][r] * l[r]);
        }
}

// ---------------------------------------------------------------------------
// Workspace layout (bytes)
// ---------------------------------------------------------------------------
static constexpr size_t OFF_SEQB = 0;                                   // 2560*2048 bf16
static constexpr size_t OFF_WQ   = OFF_SEQB + (size_t)SEQN * CN * 2;    // transposed bf16
static constexpr size_t OFF_WK   = OFF_WQ   + (size_t)CN * CN * 2;
static constexpr size_t OFF_WV   = OFF_WK   + (size_t)CN * KVN * 2;
static constexpr size_t OFF_WO   = OFF_WV   + (size_t)CN * KVN * 2;
static constexpr size_t OFF_QF   = OFF_WO   + (size_t)CN * CN * 2;      // f32
static constexpr size_t OFF_KF   = OFF_QF   + (size_t)TN * CN * 4;
static constexpr size_t OFF_VF   = OFF_KF   + (size_t)SEQN * KVN * 4;
static constexpr size_t OFF_QB   = OFF_VF   + (size_t)SEQN * KVN * 4;   // bf16 head-major
static constexpr size_t OFF_KB   = OFF_QB   + (size_t)HN * TN * DN * 2;
static constexpr size_t OFF_VT   = OFF_KB   + (size_t)HKN * TOTN * DN * 2;
static constexpr size_t OFF_YB   = OFF_VT   + (size_t)HKN * TOTN * DN * 2;
// end = OFF_YB + TN*CN*2  ~= 79 MB total

extern "C" void kernel_launch(void* const* d_in, const int* in_sizes, int n_in,
                              void* d_out, int out_size, void* d_ws, size_t ws_size,
                              hipStream_t stream) {
    (void)in_sizes; (void)n_in; (void)out_size; (void)ws_size;

    const float* x      = (const float*)d_in[0];
    const float* stm    = (const float*)d_in[1];
    // d_in[2] = long_q: its attention rows are sliced away by y[:, -T:] -> unused
    const float* long_k = (const float*)d_in[3];
    const float* long_v = (const float*)d_in[4];
    const float* Wq     = (const float*)d_in[5];
    const float* Wk     = (const float*)d_in[6];
    const float* Wv     = (const float*)d_in[7];
    const float* Wo     = (const float*)d_in[8];
    float* out = (float*)d_out;

    char* ws = (char*)d_ws;
    unsigned short* seqb = (unsigned short*)(ws + OFF_SEQB);
    unsigned short* WqT  = (unsigned short*)(ws + OFF_WQ);
    unsigned short* WkT  = (unsigned short*)(ws + OFF_WK);
    unsigned short* WvT  = (unsigned short*)(ws + OFF_WV);
    unsigned short* WoT  = (unsigned short*)(ws + OFF_WO);
    float*          Qf   = (float*)(ws + OFF_QF);
    float*          Kf   = (float*)(ws + OFF_KF);
    float*          Vf   = (float*)(ws + OFF_VF);
    unsigned short* Qb   = (unsigned short*)(ws + OFF_QB);
    unsigned short* Kb   = (unsigned short*)(ws + OFF_KB);
    unsigned short* Vt   = (unsigned short*)(ws + OFF_VT);
    unsigned short* Yb   = (unsigned short*)(ws + OFF_YB);

    auto cvt = [&](const float* s, unsigned short* d, size_t n) {
        int n4 = (int)(n >> 2);
        k_cvt_bf16<<<(n4 + 255) / 256, 256, 0, stream>>>(s, d, n4);
    };
    auto cvt_tr = [&](const float* s, unsigned short* d, int Kdim, int Ndim) {
        k_cvt_tr<<<dim3(Ndim / 64, Kdim / 64), 256, 0, stream>>>(s, d, Kdim, Ndim);
    };
    auto gemm = [&](const unsigned short* A, const unsigned short* Bt, float* C,
                    int M, int N, int K) {
        k_gemm_bf16<<<dim3(N / BN, M / BM), 256, 0, stream>>>(A, Bt, C, M, N, K);
    };

    // 1) bf16 conversions (seq = stm ‖ x is contiguous row-major); weights transposed
    cvt(stm, seqb,                   (size_t)SN * CN);
    cvt(x,   seqb + (size_t)SN * CN, (size_t)TN * CN);
    cvt_tr(Wq, WqT, CN, CN);
    cvt_tr(Wk, WkT, CN, KVN);
    cvt_tr(Wv, WvT, CN, KVN);
    cvt_tr(Wo, WoT, CN, CN);

    // 2) projections (Q only needs the x rows — earlier queries are discarded)
    gemm(seqb + (size_t)SN * CN, WqT, Qf, TN,   CN,  CN);   // 2048x2048x2048
    gemm(seqb,                   WkT, Kf, SEQN, KVN, CN);   // 2560x512x2048
    gemm(seqb,                   WvT, Vf, SEQN, KVN, CN);

    // 3) RoPE + head-major bf16 packing
    k_rope_q <<<(TN * HN * 64) / 256,    256, 0, stream>>>(Qf, Qb);
    k_build_k<<<(HKN * TOTN * 64) / 256, 256, 0, stream>>>(long_k, Kf, Kb);
    k_build_v<<<(HKN * DN * TOTN) / 256, 256, 0, stream>>>(long_v, Vf, Vt);

    // 4) flash attention (causal over 3584 keys, GQA 16q/4kv)
    k_attn<<<dim3(HN, TN / 128), 256, 0, stream>>>(Qb, Kb, Vt, Yb);

    // 5) output projection -> fp32 out
    gemm(Yb, WoT, out, TN, CN, CN);
}
